// SDP_Improved_1812476199075
// MI455X (gfx1250) — matrix operand order repaired and checked by reading, not compiled, not run on hardware
//
#include <hip/hip_runtime.h>
#include <hip/hip_bf16.h>

// MI455X (gfx1250) wave32 + bf16 WMMA (v_wmma_f32_16x16x32_bf16, f32 accum)
typedef __bf16 bf16_t;
typedef __attribute__((ext_vector_type(16))) __bf16 v16bf;
typedef __attribute__((ext_vector_type(8)))  float  v8f;
typedef __attribute__((ext_vector_type(4)))  unsigned int u32x4;
typedef __attribute__((ext_vector_type(8)))  int i32x8;
typedef __attribute__((ext_vector_type(4)))  int i32x4;

// problem constants (from reference setup)
#define CB   8          // batch
#define CC   256        // channels (= inter)
#define CH   128
#define CW   128
#define CHW  16384      // H*W
#define CKT  2304       // C*9
#define CG   32         // groups (8 ch/group)

__device__ __forceinline__ float ldval(const float* p)  { return *p; }
__device__ __forceinline__ float ldval(const bf16_t* p) { return (float)(*p); }
__device__ __forceinline__ void  stval(float* p, float v)  { *p = v; }
__device__ __forceinline__ void  stval(bf16_t* p, float v) { *p = (bf16_t)v; }

// ---------------------------------------------------------------- small utils
__global__ void zero_f32(float* p, int n) {
  int i = blockIdx.x * blockDim.x + threadIdx.x;
  if (i < n) p[i] = 0.0f;
}

__global__ __launch_bounds__(256) void cvt_bf16(const float* __restrict__ s,
                                                bf16_t* __restrict__ d, unsigned n) {
  unsigned i = blockIdx.x * 256u + threadIdx.x;
  if (i < n) d[i] = (bf16_t)s[i];
}

// w (O,C,3,3) f32 -> w' (O, tap, C) bf16 so the WMMA B K-runs are contiguous
__global__ __launch_bounds__(256) void reorder_w(const float* __restrict__ w,
                                                 bf16_t* __restrict__ wre) {
  unsigned i = blockIdx.x * 256u + threadIdx.x;   // dest index
  if (i >= (unsigned)(CC * CKT)) return;
  unsigned o = i / CKT, r = i - o * CKT;
  unsigned tap = r >> 8, c = r & 255u;
  wre[i] = (bf16_t)w[(o * CC + c) * 9u + tap];
}

__global__ void gn_finalize(float* stats, int n, float invN, float eps) {
  int i = blockIdx.x * blockDim.x + threadIdx.x;
  if (i < n) {
    float mu  = stats[2 * i] * invN;
    float var = stats[2 * i + 1] * invN - mu * mu;
    stats[2 * i]     = mu;
    stats[2 * i + 1] = rsqrtf(var + eps);
  }
}

template <typename T>
__global__ __launch_bounds__(256) void gn_apply(
    T* data, const float* __restrict__ stats,
    const float* __restrict__ gw, const float* __restrict__ gb)
{
  unsigned i = blockIdx.x * 256u + threadIdx.x;        // < 2^25, 32-bit safe
  const unsigned c = (i >> 14) & 255u;                 // HW = 2^14
  const unsigned b = i >> 22;                          // C*HW = 2^22
  const unsigned sidx = 2u * (b * CG + (c >> 3));
  const float mu = stats[sidx], rs = stats[sidx + 1];
  stval(&data[i], (ldval(&data[i]) - mu) * rs * gw[c] + gb[c]);
}

// --------------------------------------------- conv3x3 as implicit GEMM (WMMA)
// K-order: K' = tap*256 + cin (A and B agree). One wave: 16 px x 64 oc
// (4 D-tiles share each A-fragment). Weights pre-reordered bf16 (O,tap,C).
template <typename TOUT>
__global__ __launch_bounds__(256) void conv3x3_gn_stats(
    const bf16_t* __restrict__ xin, const bf16_t* __restrict__ wre,
    TOUT* __restrict__ out, float* __restrict__ stats)
{
  const int lane = threadIdx.x & 31;
  const int wv   = threadIdx.x >> 5;
  const int tile = blockIdx.x * 8 + wv;       // 32768 waves
  const int tQ = tile & 3;                    // oc quad (64 channels)
  const int tM = tile >> 2;                   // pixel block (16 px, one image row)
  const unsigned m0  = (unsigned)tM << 4;
  const unsigned b   = m0 >> 14;
  const unsigned p0  = m0 & 16383u;
  const unsigned oc0 = (unsigned)tQ << 6;
  const bool lo = (lane < 16);
  const int  ln = lane & 15;

  const unsigned pix = p0 + ln;               // A row: M = lane&15
  const int ay = (int)(pix >> 7), ax = (int)(pix & 127u);
  const bf16_t* xb = xin + b * (unsigned)(CC * CHW);
  // per-subtile weight row (B column = lane&15), +16 elems for hi half-wave
  const bf16_t* wl0 = wre + (oc0 + 0u  + ln) * CKT + (lo ? 0 : 16);
  const bf16_t* wl1 = wre + (oc0 + 16u + ln) * CKT + (lo ? 0 : 16);
  const bf16_t* wl2 = wre + (oc0 + 32u + ln) * CKT + (lo ? 0 : 16);
  const bf16_t* wl3 = wre + (oc0 + 48u + ln) * CKT + (lo ? 0 : 16);

  v8f a0 = {0,0,0,0,0,0,0,0}, a1 = a0, a2 = a0, a3 = a0;

#pragma unroll
  for (int kh = 0; kh < 3; ++kh) {
    const int yy = ay + kh - 1;
    if ((unsigned)yy >= (unsigned)CH) continue;
#pragma unroll
    for (int kw = 0; kw < 3; ++kw) {
      const int xx = ax + kw - 1;
      const bool vld = (unsigned)xx < (unsigned)CW;     // one check per tap
      const unsigned abase = (unsigned)yy * CW + (unsigned)(xx & 127);
      const unsigned tapo = (unsigned)(kh * 3 + kw) * 256u;
      for (unsigned cbv = 0; cbv < 256u; cbv += 32u) {
        __builtin_prefetch(wl0 + tapo + cbv + 64, 0, 1);
        v16bf af;
#pragma unroll
        for (int i = 0; i < 16; ++i) {
          const unsigned cin = cbv + (unsigned)i + (lo ? 0u : 16u);
          bf16_t v = (bf16_t)0.0f;
          if (vld) v = xb[cin * (unsigned)CHW + abase];
          af[i] = v;
        }
        const v16bf b0 = *(const v16bf*)(wl0 + tapo + cbv);
        const v16bf b1 = *(const v16bf*)(wl1 + tapo + cbv);
        const v16bf b2 = *(const v16bf*)(wl2 + tapo + cbv);
        const v16bf b3 = *(const v16bf*)(wl3 + tapo + cbv);
        a0 = __builtin_amdgcn_wmma_f32_16x16x32_bf16(false, af, false, b0, (short)0, a0, false, false);
        a1 = __builtin_amdgcn_wmma_f32_16x16x32_bf16(false, af, false, b1, (short)0, a1, false, false);
        a2 = __builtin_amdgcn_wmma_f32_16x16x32_bf16(false, af, false, b2, (short)0, a2, false, false);
        a3 = __builtin_amdgcn_wmma_f32_16x16x32_bf16(false, af, false, b3, (short)0, a3, false, false);
      }
    }
  }

  // D layout: lane holds one oc column; VGPR r -> pixel p0 + (lo?0:8) + r
  const unsigned rb = lo ? 0u : 8u;
  v8f accs[4] = {a0, a1, a2, a3};
#pragma unroll
  for (int s = 0; s < 4; ++s) {
    const unsigned ocB = oc0 + (unsigned)s * 16u + ln;
    TOUT* ob = out + (b * CC + ocB) * (unsigned)CHW + p0 + rb;
    float s1 = 0.f, s2 = 0.f;
#pragma unroll
    for (int r = 0; r < 8; ++r) {
      const float v = accs[s][r];
      s1 += v; s2 += v * v;
      stval(ob + r, v);
    }
    atomicAdd(&stats[2u * (b * CG + (ocB >> 3))],     s1);
    atomicAdd(&stats[2u * (b * CG + (ocB >> 3)) + 1], s2);
  }
}

// ------------------------------------------------- per-patch cross attention
struct AttnSmem {                      // exactly 64 KB
  union {
    bf16_t Qc[256][64];                // phases 1-2 ([c][t], TDM-major)
    struct { float Sf[64][64]; bf16_t Sb[64][64]; } s;  // phases 3-4
  } u;
  bf16_t Kc[256][64];                  // [c][t]
};

#if __has_builtin(__builtin_amdgcn_tensor_load_to_lds) && __has_builtin(__builtin_amdgcn_s_wait_tensorcnt)
#define USE_TDM 1
#else
#define USE_TDM 0
#endif

#if USE_TDM
// D# for a 3D tile: x=8 (contig), y=8 (stride 128), c=256 (stride 16384), 2B elems
__device__ __forceinline__ void tdm_load_tile(const bf16_t* gsrc, unsigned ldsOff) {
  const unsigned long long ga = (unsigned long long)gsrc;
  u32x4 g0; i32x8 g1; i32x4 g2, g3;
  g0[0] = 1u;                                    // count=1, user mode
  g0[1] = ldsOff;                                // lds_addr
  g0[2] = (unsigned)(ga & 0xffffffffull);        // global_addr lo
  g0[3] = (unsigned)((ga >> 32) & 0x1ffffffull) | (2u << 30);  // hi | type=2
  g1[0] = (1 << 16);                             // data_size = 2 bytes
  g1[1] = (int)(8u << 16);                       // tensor_dim0 = 8
  g1[2] = (int)(8u << 16);                       // tensor_dim1 = 8
  g1[3] = (int)(8u << 16);                       // tile_dim0 = 8
  g1[4] = (int)(8u | (256u << 16));              // tile_dim1 = 8, tile_dim2 = 256
  g1[5] = CW;                                    // tensor_dim0_stride = 128
  g1[6] = (int)((unsigned)CHW << 16);            // tensor_dim1_stride lo16 = 16384
  g1[7] = 0;
  g2[0] = 256; g2[1] = 0; g2[2] = 0; g2[3] = 0;  // tensor_dim2 = 256
  g3[0] = 0;   g3[1] = 0; g3[2] = 0; g3[3] = 0;
#if __clang_major__ >= 23
  i32x8 g4 = {0, 0, 0, 0, 0, 0, 0, 0};
  __builtin_amdgcn_tensor_load_to_lds(g0, g1, g2, g3, g4, 0);
#else
  __builtin_amdgcn_tensor_load_to_lds(g0, g1, g2, g3, 0);
#endif
}
#endif

__global__ __launch_bounds__(256) void attn_kernel(
    const bf16_t* __restrict__ qn, const bf16_t* __restrict__ kn,
    const float* __restrict__ xlow, bf16_t* __restrict__ vmap)
{
  __shared__ AttnSmem sm;

  const unsigned n  = blockIdx.x;
  const unsigned pw = n & 15u, ph = (n >> 4) & 15u, b = n >> 8;
  const unsigned y0 = ph << 3, x0 = pw << 3;
  const int tid = threadIdx.x;
  const int lane = tid & 31, wv = tid >> 5;
  const bool lo = (lane < 16);
  const int ln = lane & 15;
  const unsigned pbase = b * (unsigned)(CC * CHW) + y0 * CW + x0;  // tile origin

#if USE_TDM
  if (wv == 0) {
    tdm_load_tile(qn + pbase, (unsigned)(unsigned long long)(void*)&sm.u.Qc[0][0]);
    tdm_load_tile(kn + pbase, (unsigned)(unsigned long long)(void*)&sm.Kc[0][0]);
    __builtin_amdgcn_s_wait_tensorcnt(0);
  }
#else
  for (int idx = tid; idx < 256 * 64; idx += 256) {
    const unsigned c = (unsigned)idx >> 6, t = (unsigned)idx & 63u;
    const unsigned g = pbase + c * CHW + (t >> 3) * CW + (t & 7u);
    sm.u.Qc[c][t] = qn[g];
    sm.Kc[c][t]  = kn[g];
  }
#endif
  __syncthreads();

  // S = Q K^T / 16 : wave handles 2 of the 16 16x16 tiles (same token rows)
  const int t0s = (wv >> 1) << 4;
  const int trow = t0s + ln;
  const int sc0 = ((wv & 1) * 2 + 0) * 16 + ln;   // B col for tile 0
  const int sc1 = ((wv & 1) * 2 + 1) * 16 + ln;   // B col for tile 1
  v8f acc0 = {0,0,0,0,0,0,0,0}, acc1 = acc0;
  for (int cbv = 0; cbv < 256; cbv += 32) {
    v16bf af, b0, b1;
#pragma unroll
    for (int i = 0; i < 16; ++i) {
      const int c = cbv + i + (lo ? 0 : 16);
      af[i] = sm.u.Qc[c][trow];
    }
    const int cK = cbv + (lo ? 0 : 16);
#pragma unroll
    for (int i = 0; i < 16; ++i) b0[i] = sm.Kc[cK + i][sc0];
#pragma unroll
    for (int i = 0; i < 16; ++i) b1[i] = sm.Kc[cK + i][sc1];
    acc0 = __builtin_amdgcn_wmma_f32_16x16x32_bf16(false, af, false, b0, (short)0, acc0, false, false);
    acc1 = __builtin_amdgcn_wmma_f32_16x16x32_bf16(false, af, false, b1, (short)0, acc1, false, false);
  }
  __syncthreads();          // Qc dead -> reuse as Sf/Sb
  const int rb = lo ? 0 : 8;
#pragma unroll
  for (int r = 0; r < 8; ++r) {
    sm.u.s.Sf[t0s + rb + r][sc0] = acc0[r] * 0.0625f;   // 1/sqrt(256)
    sm.u.s.Sf[t0s + rb + r][sc1] = acc1[r] * 0.0625f;
  }
  __syncthreads();

  // row softmax (one thread per token row), bf16 result
  if (tid < 64) {
    float mx = -3.4e38f;
    for (int s = 0; s < 64; ++s) mx = fmaxf(mx, sm.u.s.Sf[tid][s]);
    float sum = 0.f;
    for (int s = 0; s < 64; ++s) sum += __expf(sm.u.s.Sf[tid][s] - mx);
    const float inv = 1.f / sum;
    for (int s = 0; s < 64; ++s)
      sm.u.s.Sb[tid][s] = (bf16_t)(__expf(sm.u.s.Sf[tid][s] - mx) * inv);
  }
  __syncthreads();

  // O = S K : 64 tiles, 8 per wave; residual add, write bf16 vmap
#pragma unroll
  for (int q = 0; q < 8; ++q) {
    const int ot = wv * 8 + q;
    const int t0 = (ot >> 4) << 4, c0 = (ot & 15) << 4;
    v8f acc = {0,0,0,0,0,0,0,0};
#pragma unroll
    for (int sb = 0; sb < 64; sb += 32) {
      v16bf af, bf2;
#pragma unroll
      for (int i = 0; i < 16; ++i) {
        const int s = sb + i + (lo ? 0 : 16);
        af[i] = sm.u.s.Sb[t0 + ln][s];
      }
      const int sK = sb + (lo ? 0 : 16);
#pragma unroll
      for (int i = 0; i < 16; ++i) bf2[i] = sm.Kc[c0 + ln][sK + i];  // contiguous
      acc = __builtin_amdgcn_wmma_f32_16x16x32_bf16(false, af, false, bf2, (short)0, acc, false, false);
    }
    const unsigned cN = (unsigned)(c0 + ln);
#pragma unroll
    for (int r = 0; r < 8; ++r) {
      const unsigned t = (unsigned)(t0 + rb + r);
      const unsigned gi = b * (unsigned)(CC * CHW) + cN * CHW
                        + (y0 + (t >> 3)) * CW + x0 + (t & 7u);
      vmap[gi] = (bf16_t)(acc[r] + xlow[gi]);
    }
  }
}

// ---------------------------------------------------------------- host side
extern "C" void kernel_launch(void* const* d_in, const int* in_sizes, int n_in,
                              void* d_out, int out_size, void* d_ws, size_t ws_size,
                              hipStream_t stream) {
  const float* x_low  = (const float*)d_in[0];
  const float* x_high = (const float*)d_in[1];
  const float* w_q  = (const float*)d_in[2];
  const float* gq_w = (const float*)d_in[3];
  const float* gq_b = (const float*)d_in[4];
  const float* w_k  = (const float*)d_in[5];
  const float* gk_w = (const float*)d_in[6];
  const float* gk_b = (const float*)d_in[7];
  const float* w_o  = (const float*)d_in[8];
  const float* go_w = (const float*)d_in[9];
  const float* go_b = (const float*)d_in[10];
  float* out = (float*)d_out;

  const size_t mapElems = (size_t)CB * CC * CHW;        // 33,554,432
  const size_t mapBytes = mapElems * 2;                 // 64 MB (bf16)
  const size_t wBytes   = (size_t)CC * CKT * 2;         // 1.125 MB (bf16)

  char* ws = (char*)d_ws;
  bf16_t* qmap = (bf16_t*)(ws + 0 * mapBytes);
  bf16_t* kmap = (bf16_t*)(ws + 1 * mapBytes);
  bf16_t* vmap = (bf16_t*)(ws + 2 * mapBytes);
  bf16_t* xlb  = (bf16_t*)(ws + 3 * mapBytes);
  bf16_t* xhb  = (bf16_t*)(ws + 4 * mapBytes);
  bf16_t* wqre = (bf16_t*)(ws + 5 * mapBytes);
  bf16_t* wkre = (bf16_t*)(ws + 5 * mapBytes + wBytes);
  bf16_t* wore = (bf16_t*)(ws + 5 * mapBytes + 2 * wBytes);
  float*  statsQ = (float*)(ws + 5 * mapBytes + 3 * wBytes);
  float*  statsK = statsQ + CB * CG * 2;
  float*  statsO = statsK + CB * CG * 2;

  const float invN = 1.0f / (float)((CC / CG) * CHW);   // 8*16384 per group
  const float eps  = 1e-5f;
  const int ewBlocks = (int)(mapElems / 256);           // 131072

  zero_f32<<<6, 256, 0, stream>>>(statsQ, 3 * CB * CG * 2);
  cvt_bf16<<<ewBlocks, 256, 0, stream>>>(x_low,  xlb, (unsigned)mapElems);
  cvt_bf16<<<ewBlocks, 256, 0, stream>>>(x_high, xhb, (unsigned)mapElems);
  reorder_w<<<(CC * CKT + 255) / 256, 256, 0, stream>>>(w_q, wqre);
  reorder_w<<<(CC * CKT + 255) / 256, 256, 0, stream>>>(w_k, wkre);
  reorder_w<<<(CC * CKT + 255) / 256, 256, 0, stream>>>(w_o, wore);

  const int convBlocks = ((CB * CHW / 16) * (CC / 64)) / 8;  // 4096

  conv3x3_gn_stats<bf16_t><<<convBlocks, 256, 0, stream>>>(xlb, wqre, qmap, statsQ);
  conv3x3_gn_stats<bf16_t><<<convBlocks, 256, 0, stream>>>(xhb, wkre, kmap, statsK);
  gn_finalize<<<2, 256, 0, stream>>>(statsQ, 2 * CB * CG, invN, eps);
  gn_apply<bf16_t><<<ewBlocks, 256, 0, stream>>>(qmap, statsQ, gq_w, gq_b);
  gn_apply<bf16_t><<<ewBlocks, 256, 0, stream>>>(kmap, statsK, gk_w, gk_b);

  attn_kernel<<<CB * 16 * 16, 256, 0, stream>>>(qmap, kmap, x_low, vmap);

  conv3x3_gn_stats<float><<<convBlocks, 256, 0, stream>>>(vmap, wore, out, statsO);
  gn_finalize<<<1, 256, 0, stream>>>(statsO, CB * CG, invN, eps);
  gn_apply<float><<<ewBlocks, 256, 0, stream>>>(out, statsO, go_w, go_b);
}
